// Model_36988258353598
// MI455X (gfx1250) — matrix operand order repaired and checked by reading, not compiled, not run on hardware
//
#include <hip/hip_runtime.h>
#include <math.h>

// ============================================================================
// MI455X / gfx1250 implementation (round 2).
//
// Pipeline (5 kernels on `stream`):
//   k_latent : 14->16->32->16 tanh MLP on 16-node WMMA tiles, shfl+LDS
//              reduction -> deterministic per-block partials in ws
//   k_tnets  : latent mean, 5 t-nets (16->48->32->d^2, +eye), fold
//              dist_transform[16x50] -> [16x25] (sin(-x)=-sin(x))
//   k_feats  : t_params(12) -> inproj(12->16->32->48->16) -> feats, plus
//              q0,q1,k0,k1 16^4 chains -- all v_wmma_f32_16x16x32_f16
//   k_attn   : one wave32 per node (lane==neighbor, K=32), butterfly softmax,
//              attention factored OUT of linear message transforms
//   k_delta  : 48->32->16->16->48->32->6 tanh MLP via WMMA -> d_out
//
// Codegen strategy: all WMMA fragment loads are UNCONDITIONAL vector LDS
// loads (ds_load_b128): activation tiles are zero-padded (TILE_LD=64, stores
// pad output width to next layer's K chunk), weights staged TRANSPOSED
// (Wt[n][k], K padded to 32/64) so B fragments are contiguous per lane.
// tanh is branch-free: 1 - 2*rcp(exp(2x)+1).
//
// d_in layout assumption: pytree leaves in setup_inputs() insertion order,
// params' nested dicts/lists/tuples depth-first:
//   0..5 latent | 6..35 five t-nets | 36..43 inproj | 44..59 query[0..1]
//   60..75 key[0..1] | 76 head_transform[2,16,16] | 77 dist_transform[2,16,50]
//   78..89 delta | 90 means 91 cov 92 u 93 boundaries 94 sample_u
//   95 sample_ux 96 sample_uxx 97 sample_pde 98 freq[6] 99 neighbor_idx int32
// ============================================================================

#define NN 65536
#define KNB 32
#define TILE_LD 64          // halves; row stride 128B -> b128-aligned slices

typedef _Float16 half_t;
typedef half_t v16h __attribute__((ext_vector_type(16)));
typedef half_t v8h  __attribute__((ext_vector_type(8)));
typedef float  v8f  __attribute__((ext_vector_type(8)));

#define N16 (NN * 16)
// workspace layout (float element offsets); total ~33.7 MB
#define WS_T     0                      // 14: Td[4] Tu Tux[4] Tuxx[4] Tp
#define WS_FOLD  32                     // 2 heads * 400 (fd16 fs192 fc192)
#define WS_PART  1024                   // 1024 blocks * 16 latent partials
#define WS_FEATS 32768                  // N*16
#define WS_Q0    (WS_FEATS + 1 * N16)
#define WS_Q1    (WS_FEATS + 2 * N16)
#define WS_K0    (WS_FEATS + 3 * N16)
#define WS_K1    (WS_FEATS + 4 * N16)
#define WS_LG    (WS_FEATS + 5 * N16)   // N*48 local_global

// ---------------------------------------------------------------------------
// helpers
// ---------------------------------------------------------------------------
__device__ __forceinline__ float tanh_fast(float x) {
  // branch-free: saturates correctly (exp->inf => 1, exp->0 => -1)
  const float e = __expf(2.0f * x);
  return 1.0f - 2.0f * __builtin_amdgcn_rcpf(e + 1.0f);
}

__device__ __forceinline__ float wsum32(float v) {
#pragma unroll
  for (int m = 16; m > 0; m >>= 1) v += __shfl_xor(v, m, 32);
  return v;
}
__device__ __forceinline__ float wmax32(float v) {
#pragma unroll
  for (int m = 16; m > 0; m >>= 1) v = fmaxf(v, __shfl_xor(v, m, 32));
  return v;
}

__device__ __forceinline__ void copy_b(const float* __restrict__ s,
                                       float* __restrict__ d, int n, int tid, int nt) {
  for (int i = tid; i < n; i += nt) d[i] = s[i];
}

// Stage W[din][dout] (row-major, global f32) -> LDS transposed Wt[np][kp] f16,
// zero-padded. kp in {32,64}, np multiple of 16 (or any >= dout).
__device__ __forceinline__ void stage_wT(const float* __restrict__ W,
                                         half_t* __restrict__ dst,
                                         int din, int dout, int kp, int np,
                                         int tid, int nt) {
  const int tot = np * kp;
  for (int i = tid; i < tot; i += nt) {
    const int nr = i / kp, k = i % kp;
    const float v = (nr < dout && k < din) ? W[k * dout + nr] : 0.0f;
    dst[i] = (half_t)v;
  }
}

// ---------------------------------------------------------------------------
// WMMA fragment loaders per CDNA5 ISA 7.12.2 (wave32). All unconditional,
// 16B-aligned vector LDS loads (tiles/weights are zero-padded).
// A 16x32 f16: lane m (m<16) holds row m, K 0..7 | 16..23; lane m+16: 8..15 | 24..31
// B 32x16 f16: lane n (n<16) holds col n, K 0..7 | 16..23; lane n+16: 8..15 | 24..31
// C/D f32: lane l elem r = D[r + 8*(l>=16)][l&15]
// ---------------------------------------------------------------------------
__device__ __forceinline__ v16h load_A(const half_t* __restrict__ tile,
                                       int kbase, int lane) {
  const int m  = lane & 15;
  const int kh = (lane & 16) ? 8 : 0;
  const v8h* p = (const v8h*)(tile + m * TILE_LD + kbase + kh);  // 16B aligned
  const v8h lo = p[0];   // K = kh .. kh+7
  const v8h hi = p[2];   // K = 16+kh .. 16+kh+7
  return __builtin_shufflevector(lo, hi, 0, 1, 2, 3, 4, 5, 6, 7,
                                 8, 9, 10, 11, 12, 13, 14, 15);
}

__device__ __forceinline__ v16h load_Bt(const half_t* __restrict__ wt, int ldk,
                                        int kbase, int nbase, int lane) {
  const int n  = lane & 15;
  const int kh = (lane & 16) ? 8 : 0;
  const v8h* p = (const v8h*)(wt + (nbase + n) * ldk + kbase + kh);  // 16B aligned
  const v8h lo = p[0];
  const v8h hi = p[2];
  return __builtin_shufflevector(lo, hi, 0, 1, 2, 3, 4, 5, 6, 7,
                                 8, 9, 10, 11, 12, 13, 14, 15);
}

// One MLP layer on a 16-node tile: out = act(in @ W + b).
// Pads output columns up to next 32-multiple with zeros so the next layer's
// unconditional K-chunk reads stay clean even when buffers are reused.
template <int DIN, int DOUT, bool ACT>
__device__ __forceinline__ void mlp_layer(const half_t* __restrict__ wt,
                                          const float* __restrict__ bias,
                                          const half_t* __restrict__ inT,
                                          half_t* __restrict__ outT, int lane) {
  constexpr int KP    = (DIN + 31) & ~31;
  constexpr int PADTO = (DOUT + 31) & ~31;
  const int n  = lane & 15;
  const int mb = (lane & 16) ? 8 : 0;
#pragma unroll
  for (int nb = 0; nb < DOUT; nb += 16) {
    v8f c = {};
#pragma unroll
    for (int kb = 0; kb < KP; kb += 32) {
      v16h av = load_A(inT, kb, lane);
      v16h bv = load_Bt(wt, KP, kb, nb, lane);
      c = __builtin_amdgcn_wmma_f32_16x16x32_f16(false, av, false, bv, (short)0,
                                                 c, false, false);
    }
    const float bb = bias[nb + n];
#pragma unroll
    for (int r = 0; r < 8; ++r) {
      float v = c[r] + bb;
      if (ACT) v = tanh_fast(v);
      outT[(mb + r) * TILE_LD + nb + n] = (half_t)v;
    }
  }
#pragma unroll
  for (int nb = DOUT; nb < PADTO; nb += 16) {
#pragma unroll
    for (int r = 0; r < 8; ++r)
      outT[(mb + r) * TILE_LD + nb + n] = (half_t)0.0f;
  }
}

// Final (linear) 16-wide layer: f32 rows to global, optional f16 tile (+pad).
template <int DIN>
__device__ __forceinline__ void mlp_layer_out16(const half_t* __restrict__ wt,
                                                const float* __restrict__ bias,
                                                const half_t* __restrict__ inT,
                                                half_t* __restrict__ outT,
                                                float* __restrict__ gout,
                                                int row_base, int lane) {
  constexpr int KP = (DIN + 31) & ~31;
  const int n  = lane & 15;
  const int mb = (lane & 16) ? 8 : 0;
  v8f c = {};
#pragma unroll
  for (int kb = 0; kb < KP; kb += 32) {
    v16h av = load_A(inT, kb, lane);
    v16h bv = load_Bt(wt, KP, kb, 0, lane);
    c = __builtin_amdgcn_wmma_f32_16x16x32_f16(false, av, false, bv, (short)0,
                                               c, false, false);
  }
  const float bb = bias[n];
#pragma unroll
  for (int r = 0; r < 8; ++r) {
    const float v = c[r] + bb;
    gout[(size_t)(row_base + mb + r) * 16 + n] = v;
    if (outT) outT[(mb + r) * TILE_LD + n] = (half_t)v;
  }
  if (outT) {
#pragma unroll
    for (int r = 0; r < 8; ++r)
      outT[(mb + r) * TILE_LD + 16 + n] = (half_t)0.0f;
  }
}

// ---------------------------------------------------------------------------
// Kernel 1: latent MLP (tanh on ALL layers) on WMMA tiles + reduction.
// 1024 blocks x 128 threads (4 waves, one 16-node tile each).
// ---------------------------------------------------------------------------
struct K1Args {
  const float *W0, *B0, *W1, *B1, *W2, *B2;
  const float *means, *cov, *u, *bnd, *su, *sux, *suxx, *spde;
  float* part;
};

__global__ void __launch_bounds__(128) k_latent(K1Args a) {
  __shared__ alignas(16) half_t wl[2048];
  __shared__ float bl[64];
  __shared__ alignas(16) half_t tl[4 * 2 * 16 * TILE_LD];
  __shared__ float red[4][16];
  const int tid = threadIdx.x;
  {
    unsigned* z = (unsigned*)tl;
    for (int i = tid; i < 4096; i += 128) z[i] = 0u;
  }
  stage_wT(a.W0, wl + 0,    14, 16, 32, 16, tid, 128);
  stage_wT(a.W1, wl + 512,  16, 32, 32, 32, tid, 128);
  stage_wT(a.W2, wl + 1536, 32, 16, 32, 16, tid, 128);
  copy_b(a.B0, bl + 0, 16, tid, 128);
  copy_b(a.B1, bl + 16, 32, tid, 128);
  copy_b(a.B2, bl + 48, 16, tid, 128);
  __syncthreads();

  const int wv = tid >> 5, ln = tid & 31;
  half_t* bufX = tl + (wv * 2 + 0) * 16 * TILE_LD;
  half_t* bufY = tl + (wv * 2 + 1) * 16 * TILE_LD;
  const int tb = blockIdx.x * 64 + wv * 16;

  if (ln < 16) {  // inp row (14 features)
    const int n = tb + ln;
    half_t* r = bufX + ln * TILE_LD;
    r[0] = (half_t)a.means[2 * n];     r[1] = (half_t)a.means[2 * n + 1];
    r[2] = (half_t)a.cov[4 * n];       r[3] = (half_t)a.cov[4 * n + 1];
    r[4] = (half_t)a.cov[4 * n + 2];   r[5] = (half_t)a.cov[4 * n + 3];
    r[6] = (half_t)a.u[n];             r[7] = (half_t)a.bnd[n];
    r[8] = (half_t)a.su[n];
    r[9] = (half_t)a.sux[2 * n];       r[10] = (half_t)a.sux[2 * n + 1];
    r[11] = (half_t)a.suxx[2 * n];     r[12] = (half_t)a.suxx[2 * n + 1];
    r[13] = (half_t)a.spde[n];
  }
  mlp_layer<14, 16, true>(wl + 0,   bl + 0,  bufX, bufY, ln);
  mlp_layer<16, 32, true>(wl + 512, bl + 16, bufY, bufX, ln);
  {  // final 32->16, tanh (act_all=True), reduce over the 16 tile rows
    v16h av = load_A(bufX, 0, ln);
    v16h bv = load_Bt(wl + 1536, 32, 0, 0, ln);
    v8f c = {};
    c = __builtin_amdgcn_wmma_f32_16x16x32_f16(false, av, false, bv, (short)0,
                                               c, false, false);
    const int n = ln & 15;
    const float bb = bl[48 + n];
    float p = 0.f;
#pragma unroll
    for (int r = 0; r < 8; ++r) p += tanh_fast(c[r] + bb);
    p += __shfl_xor(p, 16, 32);  // rows 0..7 + rows 8..15 of column n
    if (ln < 16) red[wv][ln] = p;
  }
  __syncthreads();
  if (tid < 16) {
    const float s = red[0][tid] + red[1][tid] + red[2][tid] + red[3][tid];
    a.part[blockIdx.x * 16 + tid] = s;  // fixed order => deterministic
  }
}

// ---------------------------------------------------------------------------
// Kernel 2: latent mean, t-nets (+eye), fold dist_transform 50 -> 25
// ---------------------------------------------------------------------------
struct TnetW { const float *W0, *B0, *W1, *B1, *W2, *B2; int dd; int off; };
struct TnetPack { TnetW t[5]; };

__global__ void __launch_bounds__(256)
k_tnets(float* __restrict__ ws, TnetPack pack, const float* __restrict__ distT) {
  __shared__ float lat[16], h1[48], h2[32];
  const int tid = threadIdx.x;
  if (tid < 16) {
    float s = 0.f;
    for (int b = 0; b < 1024; ++b) s += ws[WS_PART + b * 16 + tid];
    lat[tid] = s * (1.0f / 65536.0f);
  }
  __syncthreads();
  for (int tn = 0; tn < 5; ++tn) {
    const TnetW T = pack.t[tn];
    if (tid < 48) {
      float s = T.B0[tid];
      for (int i = 0; i < 16; ++i) s += lat[i] * T.W0[i * 48 + tid];
      h1[tid] = tanh_fast(s);
    }
    __syncthreads();
    if (tid < 32) {
      float s = T.B1[tid];
      for (int i = 0; i < 48; ++i) s += h1[i] * T.W1[i * 32 + tid];
      h2[tid] = tanh_fast(s);
    }
    __syncthreads();
    if (tid < T.dd) {
      float s = T.B2[tid];
      for (int i = 0; i < 32; ++i) s += h2[i] * T.W2[i * T.dd + tid];
      const int d = (T.dd == 4) ? 2 : 1;
      if ((tid / d) == (tid % d)) s += 1.0f;  // + eye(d)
      ws[WS_T + T.off + tid] = s;
    }
    __syncthreads();
  }
  // fold: emb2 = [dist,sin,cos, dist,-sin,cos]
  for (int p = tid; p < 32; p += 256) {
    const int h = p >> 4, i = p & 15;
    const float* row = distT + h * 800 + i * 50;
    ws[WS_FOLD + h * 400 + i] = row[0] + row[25];
  }
  for (int p = tid; p < 384; p += 256) {
    const int h = p / 192, q = p % 192, i = q / 12, j = q % 12;
    const float* row = distT + h * 800 + i * 50;
    ws[WS_FOLD + h * 400 + 16 + q]  = row[1 + j]  - row[26 + j];  // sin coeff
    ws[WS_FOLD + h * 400 + 208 + q] = row[13 + j] + row[38 + j];  // cos coeff
  }
}

// ---------------------------------------------------------------------------
// Kernel 3: t_params -> inproj -> feats; q0,q1,k0,k1 chains. WMMA tiles.
// 1024 blocks x 128 threads (4 waves, one 16-node tile each).
// ---------------------------------------------------------------------------
struct K3Args {
  const float* ipW[4]; const float* ipB[4];
  const float* qkW[4][4]; const float* qkB[4][4];  // m: q0,q1,k0,k1
  const float *cov, *u, *bnd, *su, *sux, *suxx, *spde;
  float* ws;
};

__global__ void __launch_bounds__(128) k_feats(K3Args a) {
  __shared__ alignas(16) half_t wl[12288];  // transposed padded weights
  __shared__ float bl[368];
  __shared__ alignas(16) half_t tl[4 * 3 * 16 * TILE_LD];
  const int tid = threadIdx.x;
  {
    unsigned* z = (unsigned*)tl;
    for (int i = tid; i < 6144; i += 128) z[i] = 0u;
  }
  stage_wT(a.ipW[0], wl + 0,    12, 16, 32, 16, tid, 128);
  stage_wT(a.ipW[1], wl + 512,  16, 32, 32, 32, tid, 128);
  stage_wT(a.ipW[2], wl + 1536, 32, 48, 32, 48, tid, 128);
  stage_wT(a.ipW[3], wl + 3072, 48, 16, 64, 16, tid, 128);
  copy_b(a.ipB[0], bl + 0, 16, tid, 128);
  copy_b(a.ipB[1], bl + 16, 32, tid, 128);
  copy_b(a.ipB[2], bl + 48, 48, tid, 128);
  copy_b(a.ipB[3], bl + 96, 16, tid, 128);
  for (int m = 0; m < 4; ++m)
    for (int l = 0; l < 4; ++l) {
      stage_wT(a.qkW[m][l], wl + 4096 + (m * 4 + l) * 512, 16, 16, 32, 16, tid, 128);
      copy_b(a.qkB[m][l], bl + 112 + m * 64 + l * 16, 16, tid, 128);
    }
  __syncthreads();

  const int wv = tid >> 5, ln = tid & 31;
  half_t* bufX = tl + (wv * 3 + 0) * 16 * TILE_LD;
  half_t* bufY = tl + (wv * 3 + 1) * 16 * TILE_LD;
  half_t* bufF = tl + (wv * 3 + 2) * 16 * TILE_LD;
  const int tb = blockIdx.x * 64 + wv * 16;
  float* ws = a.ws;

  if (ln < 16) {  // per-node t_params (12) -> A tile row
    const int n = tb + ln;
    const float Td00 = ws[0], Td01 = ws[1], Td10 = ws[2], Td11 = ws[3];
    const float Tu = ws[4];
    const float Ux00 = ws[5], Ux01 = ws[6], Ux10 = ws[7], Ux11 = ws[8];
    const float Xx00 = ws[9], Xx01 = ws[10], Xx10 = ws[11], Xx11 = ws[12];
    const float Tp = ws[13];
    const float c00 = a.cov[4 * n], c01 = a.cov[4 * n + 1];
    const float c10 = a.cov[4 * n + 2], c11 = a.cov[4 * n + 3];
    half_t* r = bufX + ln * TILE_LD;
    r[0] = (half_t)(Td00 * c00 + Td01 * c10);
    r[1] = (half_t)(Td00 * c01 + Td01 * c11);
    r[2] = (half_t)(Td10 * c00 + Td11 * c10);
    r[3] = (half_t)(Td10 * c01 + Td11 * c11);
    r[4] = (half_t)(a.u[n] * Tu);
    r[5] = (half_t)a.bnd[n];
    r[6] = (half_t)(a.su[n] * Tu);
    const float ux0 = a.sux[2 * n], ux1 = a.sux[2 * n + 1];
    r[7] = (half_t)(ux0 * Ux00 + ux1 * Ux01);
    r[8] = (half_t)(ux0 * Ux10 + ux1 * Ux11);
    const float xx0 = a.suxx[2 * n], xx1 = a.suxx[2 * n + 1];
    r[9]  = (half_t)(xx0 * Xx00 + xx1 * Xx01);
    r[10] = (half_t)(xx0 * Xx10 + xx1 * Xx11);
    r[11] = (half_t)(a.spde[n] * Tp);
  }

  // inproj: 12->16->32->48->16 (last linear) -> feats (f32) + bufF (f16)
  mlp_layer<12, 16, true>(wl + 0,    bl + 0,  bufX, bufY, ln);
  mlp_layer<16, 32, true>(wl + 512,  bl + 16, bufY, bufX, ln);
  mlp_layer<32, 48, true>(wl + 1536, bl + 48, bufX, bufY, ln);
  mlp_layer_out16<48>(wl + 3072, bl + 96, bufY, bufF, ws + WS_FEATS, tb, ln);

  const int gq[4] = {WS_Q0, WS_Q1, WS_K0, WS_K1};
#pragma unroll
  for (int m = 0; m < 4; ++m) {
    const half_t* wb = wl + 4096 + m * 2048;
    const float*  bb = bl + 112 + m * 64;
    mlp_layer<16, 16, true>(wb + 0,    bb + 0,  bufF, bufX, ln);
    mlp_layer<16, 16, true>(wb + 512,  bb + 16, bufX, bufY, ln);
    mlp_layer<16, 16, true>(wb + 1024, bb + 32, bufY, bufX, ln);
    mlp_layer_out16<16>(wb + 1536, bb + 48, bufX, nullptr, ws + gq[m], tb, ln);
  }
}

// ---------------------------------------------------------------------------
// Kernel 4: attention, one wave per node, lane == neighbor (K == wave32)
// ---------------------------------------------------------------------------
__global__ void __launch_bounds__(256)
k_attn(const float* __restrict__ means, const int* __restrict__ nidx,
       const float* __restrict__ freq, const float* __restrict__ headT,
       float* __restrict__ ws) {
  const int wv = threadIdx.x >> 5, ln = threadIdx.x & 31;
  const int n = blockIdx.x * 8 + wv;
  const float* feats = ws + WS_FEATS;

  const float fx = means[2 * n], fy = means[2 * n + 1];
  const int nb = nidx[n * KNB + ln];
  const float dx = means[2 * nb] - fx, dy = means[2 * nb + 1] - fy;
  const float dist = sqrtf(dx * dx + dy * dy + 1e-12f);
  float sn[12], cs[12];
#pragma unroll
  for (int j = 0; j < 6; ++j) {
    const float f = freq[j];
    __sincosf(dx * f, &sn[j], &cs[j]);
    __sincosf(dy * f, &sn[6 + j], &cs[6 + j]);
  }
  float fnb[16];
#pragma unroll
  for (int f = 0; f < 16; ++f) fnb[f] = feats[nb * 16 + f];
  __builtin_prefetch(&ws[WS_K1 + (size_t)nb * 16], 0, 0);  // head-1 keys

#pragma unroll
  for (int h = 0; h < 2; ++h) {
    const float* qh = ws + (h ? WS_Q1 : WS_Q0) + (size_t)n * 16;
    const float* kh = ws + (h ? WS_K1 : WS_K0);
    float s = 0.f;
#pragma unroll
    for (int f = 0; f < 16; ++f) s += qh[f] * kh[(size_t)nb * 16 + f];
    s *= 0.25f;  // 1/sqrt(16)
    const float mx = wmax32(s);
    const float e = __expf(s - mx);
    const float at = e * __builtin_amdgcn_rcpf(wsum32(e));

    float acc[41];
#pragma unroll
    for (int f = 0; f < 16; ++f) acc[f] = wsum32(at * fnb[f]);
    acc[16] = wsum32(at * dist);
#pragma unroll
    for (int j = 0; j < 12; ++j) {
      acc[17 + j] = wsum32(at * sn[j]);
      acc[29 + j] = wsum32(at * cs[j]);
    }
    if (ln < 16) {
      const float* H  = headT + h * 256 + ln * 16;
      const float* fo = ws + WS_FOLD + h * 400;
      float o = 0.f;
#pragma unroll
      for (int j = 0; j < 16; ++j) o += H[j] * acc[j];
      o += fo[ln] * acc[16];
#pragma unroll
      for (int j = 0; j < 12; ++j)
        o += fo[16 + ln * 12 + j] * acc[17 + j] + fo[208 + ln * 12 + j] * acc[29 + j];
      ws[WS_LG + (size_t)n * 48 + 16 + h * 16 + ln] = o;
    }
  }
  if (ln < 16) ws[WS_LG + (size_t)n * 48 + ln] = feats[(size_t)n * 16 + ln];
}

// ---------------------------------------------------------------------------
// Kernel 5: delta MLP 48->32->16->16->48->32->6 via WMMA -> d_out
// 512 blocks x 256 threads (8 waves, one 16-node tile each).
// ---------------------------------------------------------------------------
struct K4bArgs { const float* W[6]; const float* B[6]; const float* lg; float* out; };

__global__ void __launch_bounds__(256) k_delta(K4bArgs a) {
  __shared__ alignas(16) half_t wl[7168];
  __shared__ float bl[150];
  __shared__ alignas(16) half_t tl[8 * 2 * 16 * TILE_LD];
  const int tid = threadIdx.x;
  {
    unsigned* z = (unsigned*)tl;
    for (int i = tid; i < 8192; i += 256) z[i] = 0u;
  }
  stage_wT(a.W[0], wl + 0,    48, 32, 64, 32, tid, 256);
  stage_wT(a.W[1], wl + 2048, 32, 16, 32, 16, tid, 256);
  stage_wT(a.W[2], wl + 2560, 16, 16, 32, 16, tid, 256);
  stage_wT(a.W[3], wl + 3072, 16, 48, 32, 48, tid, 256);
  stage_wT(a.W[4], wl + 4608, 48, 32, 64, 32, tid, 256);
  stage_wT(a.W[5], wl + 6656, 32, 6, 32, 16, tid, 256);
  copy_b(a.B[0], bl + 0, 32, tid, 256);
  copy_b(a.B[1], bl + 32, 16, tid, 256);
  copy_b(a.B[2], bl + 48, 16, tid, 256);
  copy_b(a.B[3], bl + 64, 48, tid, 256);
  copy_b(a.B[4], bl + 112, 32, tid, 256);
  copy_b(a.B[5], bl + 144, 6, tid, 256);
  __syncthreads();

  const int wv = tid >> 5, ln = tid & 31;
  half_t* bufX = tl + (wv * 2 + 0) * 16 * TILE_LD;
  half_t* bufY = tl + (wv * 2 + 1) * 16 * TILE_LD;
  const int tb = blockIdx.x * 128 + wv * 16;

  for (int i = ln; i < 16 * 48; i += 32) {  // load local_global tile (48 cols)
    const int r = i / 48, c = i % 48;
    bufX[r * TILE_LD + c] = (half_t)a.lg[(size_t)(tb + r) * 48 + c];
  }

  mlp_layer<48, 32, true>(wl + 0,    bl + 0,   bufX, bufY, ln);
  mlp_layer<32, 16, true>(wl + 2048, bl + 32,  bufY, bufX, ln);
  mlp_layer<16, 16, true>(wl + 2560, bl + 48,  bufX, bufY, ln);
  mlp_layer<16, 48, true>(wl + 3072, bl + 64,  bufY, bufX, ln);
  mlp_layer<48, 32, true>(wl + 4608, bl + 112, bufX, bufY, ln);
  {  // final 32->6 (weights padded to 16 cols with zeros), linear -> global
    v16h av = load_A(bufY, 0, ln);
    v16h bv = load_Bt(wl + 6656, 32, 0, 0, ln);
    v8f c = {};
    c = __builtin_amdgcn_wmma_f32_16x16x32_f16(false, av, false, bv, (short)0,
                                               c, false, false);
    const int nn = ln & 15, mb = (ln & 16) ? 8 : 0;
    if (nn < 6) {
      const float bb = bl[144 + nn];
#pragma unroll
      for (int r = 0; r < 8; ++r)
        a.out[(size_t)(tb + mb + r) * 6 + nn] = c[r] + bb;
    }
  }
}

// ---------------------------------------------------------------------------
// Host launcher
// ---------------------------------------------------------------------------
extern "C" void kernel_launch(void* const* d_in, const int* in_sizes, int n_in,
                              void* d_out, int out_size, void* d_ws, size_t ws_size,
                              hipStream_t stream) {
  (void)in_sizes; (void)out_size; (void)ws_size;
  const float* P[100];
  const int np = (n_in < 100) ? n_in : 100;
  for (int i = 0; i < np; ++i) P[i] = (const float*)d_in[i];
  float* ws  = (float*)d_ws;
  float* out = (float*)d_out;

  // 1) latent partials (WMMA tiles)
  K1Args a1;
  a1.W0 = P[0]; a1.B0 = P[1]; a1.W1 = P[2]; a1.B1 = P[3]; a1.W2 = P[4]; a1.B2 = P[5];
  a1.means = P[90]; a1.cov = P[91]; a1.u = P[92]; a1.bnd = P[93];
  a1.su = P[94]; a1.sux = P[95]; a1.suxx = P[96]; a1.spde = P[97];
  a1.part = ws + WS_PART;
  k_latent<<<1024, 128, 0, stream>>>(a1);

  // 2) t-nets + fold
  TnetPack tp;
  const int toff[5]  = {0, 4, 5, 9, 13};
  const int tdd[5]   = {4, 1, 4, 4, 1};
  const int tbase[5] = {6, 12, 18, 24, 30};
  for (int t = 0; t < 5; ++t) {
    tp.t[t].W0 = P[tbase[t] + 0]; tp.t[t].B0 = P[tbase[t] + 1];
    tp.t[t].W1 = P[tbase[t] + 2]; tp.t[t].B1 = P[tbase[t] + 3];
    tp.t[t].W2 = P[tbase[t] + 4]; tp.t[t].B2 = P[tbase[t] + 5];
    tp.t[t].dd = tdd[t]; tp.t[t].off = toff[t];
  }
  k_tnets<<<1, 256, 0, stream>>>(ws, tp, P[77]);

  // 3) feats + q/k (WMMA)
  K3Args a3;
  for (int l = 0; l < 4; ++l) { a3.ipW[l] = P[36 + 2 * l]; a3.ipB[l] = P[37 + 2 * l]; }
  for (int m = 0; m < 4; ++m)
    for (int l = 0; l < 4; ++l) {
      a3.qkW[m][l] = P[44 + m * 8 + 2 * l];
      a3.qkB[m][l] = P[45 + m * 8 + 2 * l];
    }
  a3.cov = P[91]; a3.u = P[92]; a3.bnd = P[93]; a3.su = P[94];
  a3.sux = P[95]; a3.suxx = P[96]; a3.spde = P[97]; a3.ws = ws;
  k_feats<<<1024, 128, 0, stream>>>(a3);

  // 4) attention (wave per node)
  k_attn<<<NN / 8, 256, 0, stream>>>(P[90], (const int*)d_in[99], P[98], P[76], ws);

  // 5) delta MLP (WMMA) -> out
  K4bArgs a4;
  for (int l = 0; l < 6; ++l) { a4.W[l] = P[78 + 2 * l]; a4.B[l] = P[79 + 2 * l]; }
  a4.lg = ws + WS_LG; a4.out = out;
  k_delta<<<512, 256, 0, stream>>>(a4);
}
